// SANet_33389075759210
// MI455X (gfx1250) — compile-verified, matrix operand order checked by reading, not run on hardware
//
#include <hip/hip_runtime.h>
#include <math.h>

// ---------------------------------------------------------------------------
// SANet forward for MI455X (gfx1250, wave32).
// All GEMM-shaped ops run on V_WMMA_F32_16X16X32_BF16 (bf16 in, f32 accum).
// Activations staged pixel-major [N,128] bf16; weights pre-cast to bf16 [OC,K].
// Dual accumulators per K-loop avoid WMMA->WMMA RAW hazard bubbles.
// ---------------------------------------------------------------------------

typedef __attribute__((ext_vector_type(16))) __bf16 v16bf;
typedef __attribute__((ext_vector_type(8)))  __bf16 v8bf;
typedef __attribute__((ext_vector_type(8)))  float  v8f;

#define WMMA_BF16(acc, A, Bf) \
  (acc) = __builtin_amdgcn_wmma_f32_16x16x32_bf16(false, (A), false, (Bf), (short)0, (acc), false, false)

__device__ __forceinline__ v8f zero8() {
  v8f z;
#pragma unroll
  for (int i = 0; i < 8; ++i) z[i] = 0.0f;
  return z;
}

__device__ __forceinline__ float fast_rcp(float x) { return __builtin_amdgcn_rcpf(x); }
__device__ __forceinline__ float fast_sigmoid(float x) {
  return __builtin_amdgcn_rcpf(1.0f + __expf(-x));
}

// Load a 16x32 bf16 fragment from a row-major [rows, ld] buffer.
// Per ISA 7.12.2: lanes 0-15 row=lane, K=0..7 & 16..23; lanes 16-31 same rows,
// K=8..15 & 24..31.  Used for both A (row=M) and B (row=N, transposed view).
__device__ __forceinline__ v16bf load_frag(const __bf16* tile, int ld) {
  const int lane = threadIdx.x & 31;
  const int r = lane & 15, hi = lane >> 4;
  const __bf16* p = tile + r * ld + hi * 8;
  v8bf lo = *(const v8bf*)p;
  v8bf hh = *(const v8bf*)(p + 16);
  v16bf f;
#pragma unroll
  for (int i = 0; i < 8; ++i) { f[i] = lo[i]; f[i + 8] = hh[i]; }
  return f;
}

// B fragment for patch-embed: patch n -> (b,ph,pw); k = (ky*4+kx)*128 + ic.
__device__ __forceinline__ v16bf load_frag_pe(const __bf16* sT, int p0, int k0) {
  const int lane = threadIdx.x & 31;
  const int n = p0 + (lane & 15);
  const int hi = lane >> 4;
  const int b = n >> 10, ph = (n >> 5) & 31, pw = n & 31;
  const int pp = k0 >> 7, ky = pp >> 2, kx = pp & 3;
  const int y = ph * 4 + ky, x = pw * 4 + kx;
  const __bf16* p = sT + ((size_t)((b << 14) + (y << 7) + x) << 7) + (k0 & 127) + hi * 8;
  v8bf lo = *(const v8bf*)p, hh = *(const v8bf*)(p + 16);
  v16bf f;
#pragma unroll
  for (int i = 0; i < 8; ++i) { f[i] = lo[i]; f[i + 8] = hh[i]; }
  return f;
}

// B fragment for 3x3 SAME conv: k = (ky*3+kx)*128 + ic, zero padding at borders.
__device__ __forceinline__ v16bf load_frag_3x3(const __bf16* xT, int n0, int k0) {
  const int lane = threadIdx.x & 31;
  const int n = n0 + (lane & 15);
  const int hi = lane >> 4;
  const int b = n >> 14, y = (n >> 7) & 127, x = n & 127;
  const int pp = k0 >> 7;
  const int ky = pp / 3 - 1, kx = pp - (pp / 3) * 3 - 1;
  const int yy = y + ky, xx = x + kx;
  v16bf f;
  if ((unsigned)yy < 128u && (unsigned)xx < 128u) {
    const __bf16* p = xT + ((size_t)((b << 14) + (yy << 7) + xx) << 7) + (k0 & 127) + hi * 8;
    v8bf lo = *(const v8bf*)p, hh = *(const v8bf*)(p + 16);
#pragma unroll
    for (int i = 0; i < 8; ++i) { f[i] = lo[i]; f[i + 8] = hh[i]; }
  } else {
#pragma unroll
    for (int i = 0; i < 16; ++i) f[i] = (__bf16)0.0f;
  }
  return f;
}

// ---------------------------------------------------------------------------
// Weight prep: f32 -> bf16 with K-reordering. 626688 elements total.
// ---------------------------------------------------------------------------
__global__ void k_prep_weights(const float* sm_w, const float* c1_w, const float* c2_w,
                               const float* pe_w, const float* ex_w, const float* up_w,
                               const float* pre_w, const float* post_w, __bf16* out) {
  int i = blockIdx.x * blockDim.x + threadIdx.x;
  if (i >= 626688) return;
  float v;
  if (i < 81920) {                    // sm [640,128]
    v = sm_w[i];
  } else if (i < 122880) {            // c1 [64,640]
    v = c1_w[i - 81920];
  } else if (i < 131072) {            // c2 [64,128]
    v = c2_w[i - 122880];
  } else if (i < 393216) {            // pe [128, 2048] with k=(ky*4+kx)*128+ic
    int l = i - 131072; int oc = l >> 11; int rest = l & 2047;
    int kxy = rest >> 7; int ic = rest & 127;
    v = pe_w[oc * 2048 + ic * 16 + kxy];
  } else if (i < 458752) {            // ex^T [512,128]
    int l = i - 393216; int j = l >> 7; int c = l & 127;
    v = ex_w[c * 512 + j];
  } else if (i < 462848) {            // up [128,32]
    v = up_w[i - 458752];
  } else if (i < 479232) {            // pre [128,128]
    v = pre_w[i - 462848];
  } else {                            // post [128, 1152] with k=(ky*3+kx)*128+ic
    int l = i - 479232; int oc = l / 1152; int rest = l - oc * 1152;
    int kxy = rest >> 7; int ic = rest & 127;
    v = post_w[oc * 1152 + ic * 9 + kxy];
  }
  out[i] = (__bf16)v;
}

__global__ void k_zero_acc(float* acc) {
  for (int i = threadIdx.x; i < 774; i += 256) acc[i] = 0.0f;
}

// ---------------------------------------------------------------------------
// K1: Sn = l2norm(Structure, axis=1) -> snT [N,128] bf16;  queryT bf16 copy.
// Thread-per-pixel: lanes hit consecutive pixels -> coalesced NCHW reads.
// ---------------------------------------------------------------------------
__global__ void k_norm_transpose(const float* S, const float* Q, __bf16* snT, __bf16* qT) {
  int n = blockIdx.x * 256 + threadIdx.x;
  int b = n >> 14, pix = n & 16383;
  const float* sp = S + (size_t)b * 2097152 + pix;
  const float* qp = Q + (size_t)b * 2097152 + pix;
  float ss = 0.f;
  for (int c = 0; c < 128; ++c) { float v = sp[(size_t)c * 16384]; ss += v * v; }
  float inv = fast_rcp(fmaxf(sqrtf(ss), 1e-12f));
  __bf16* so = snT + (size_t)n * 128;
  __bf16* qo = qT + (size_t)n * 128;
  for (int c = 0; c < 128; ++c) {
    so[c] = (__bf16)(sp[(size_t)c * 16384] * inv);
    qo[c] = (__bf16)(qp[(size_t)c * 16384]);
  }
}

// ---------------------------------------------------------------------------
// K2: fused read stage.  mod = sm(Sn); m0 = m_items * sigmoid(mod) (in LDS);
// s_n = concat(c1(m0), c2(Sn)) -> s_nT [N,128] bf16.   32 pixels / block.
// ---------------------------------------------------------------------------
__global__ __launch_bounds__(256) void k_read_fused(
    const __bf16* snT, const __bf16* wA_sm, const __bf16* wA_c1, const __bf16* wA_c2,
    const float* sm_b, const float* c1_b, const float* c2_b, const float* m_items,
    __bf16* s_nT) {
  __shared__ __bf16 lds_m0[32][648];   // padded rows: conflict-free 16B ds ops
  const int wave = threadIdx.x >> 5;
  const int lane = threadIdx.x & 31;
  const int r = lane & 15, hi = lane >> 4;
  const int pixbase = blockIdx.x * 32;

  // stage A: mod tile [640 x 32px], sigmoid * m_items -> LDS bf16
  for (int t = wave; t < 80; t += 8) {
    int px = t & 1, oct = t >> 1;            // oct 0..39
    int n0 = pixbase + px * 16;
    v8f acc0 = zero8(), acc1 = zero8();
#pragma unroll
    for (int kb = 0; kb < 4; kb += 2) {
      v16bf a0 = load_frag(wA_sm + oct * 16 * 128 + kb * 32, 128);
      v16bf b0 = load_frag(snT + (size_t)n0 * 128 + kb * 32, 128);
      v16bf a1 = load_frag(wA_sm + oct * 16 * 128 + (kb + 1) * 32, 128);
      v16bf b1 = load_frag(snT + (size_t)n0 * 128 + (kb + 1) * 32, 128);
      WMMA_BF16(acc0, a0, b0);
      WMMA_BF16(acc1, a1, b1);
    }
    v8f acc = acc0 + acc1;
    int n_loc = px * 16 + r;
    int cb = oct * 16 + hi * 8;
    v8bf pk;
#pragma unroll
    for (int e = 0; e < 8; ++e) {
      int j = cb + e;
      float sg = fast_sigmoid(acc[e] + sm_b[j]);
      pk[e] = (__bf16)(m_items[j] * sg);
    }
    *(v8bf*)&lds_m0[n_loc][cb] = pk;
  }
  __syncthreads();

  // stage B: m0c = c1(m0), 64 ch
  {
    int px = wave & 1, oct = wave >> 1;      // oct 0..3
    int n0 = pixbase + px * 16;
    v8f acc0 = zero8(), acc1 = zero8();
    for (int kb = 0; kb < 20; kb += 2) {
      v16bf a0 = load_frag(wA_c1 + oct * 16 * 640 + kb * 32, 640);
      v16bf b0 = load_frag(&lds_m0[px * 16][0] + kb * 32, 648);
      v16bf a1 = load_frag(wA_c1 + oct * 16 * 640 + (kb + 1) * 32, 640);
      v16bf b1 = load_frag(&lds_m0[px * 16][0] + (kb + 1) * 32, 648);
      WMMA_BF16(acc0, a0, b0);
      WMMA_BF16(acc1, a1, b1);
    }
    v8f acc = acc0 + acc1;
    int cb = oct * 16 + hi * 8;
    v8bf pk;
#pragma unroll
    for (int e = 0; e < 8; ++e) pk[e] = (__bf16)(acc[e] + c1_b[cb + e]);
    *(v8bf*)(s_nT + (size_t)(n0 + r) * 128 + cb) = pk;
  }
  // stage C: snc = c2(Sn), 64 ch (channels 64..127)
  {
    int px = wave & 1, oct = wave >> 1;
    int n0 = pixbase + px * 16;
    v8f acc0 = zero8(), acc1 = zero8();
#pragma unroll
    for (int kb = 0; kb < 4; kb += 2) {
      v16bf a0 = load_frag(wA_c2 + oct * 16 * 128 + kb * 32, 128);
      v16bf b0 = load_frag(snT + (size_t)n0 * 128 + kb * 32, 128);
      v16bf a1 = load_frag(wA_c2 + oct * 16 * 128 + (kb + 1) * 32, 128);
      v16bf b1 = load_frag(snT + (size_t)n0 * 128 + (kb + 1) * 32, 128);
      WMMA_BF16(acc0, a0, b0);
      WMMA_BF16(acc1, a1, b1);
    }
    v8f acc = acc0 + acc1;
    int cb = oct * 16 + hi * 8;
    v8bf pk;
#pragma unroll
    for (int e = 0; e < 8; ++e) pk[e] = (__bf16)(acc[e] + c2_b[cb + e]);
    *(v8bf*)(s_nT + (size_t)(n0 + r) * 128 + 64 + cb) = pk;
  }
}

// ---------------------------------------------------------------------------
// K3: patch-embed conv (4x4 stride4, K=2048) + bias + LayerNorm -> fT [4096,128]
// ---------------------------------------------------------------------------
__global__ __launch_bounds__(256) void k_patch_embed(
    const __bf16* s_nT, const __bf16* wA_pe, const float* pe_b,
    const float* ln_g, const float* ln_b, __bf16* fT) {
  __shared__ float lds_f[32][132];
  __shared__ float mu[32], rs[32];
  const int wave = threadIdx.x >> 5, lane = threadIdx.x & 31;
  const int r = lane & 15, hi = lane >> 4;
  const int pbase = blockIdx.x * 32;
  for (int t = wave; t < 16; t += 8) {
    int px = t & 1, oct = t >> 1;            // oct 0..7
    int p0 = pbase + px * 16;
    const __bf16* wrow = wA_pe + oct * 16 * 2048;
    v8f acc0 = zero8(), acc1 = zero8();
    for (int kb = 0; kb < 64; kb += 2) {
      if ((kb & 7) == 0)                       // gfx1250 prefetch of A stream
        __builtin_prefetch(wrow + (kb + 8) * 32, 0, 0);
      v16bf a0 = load_frag(wrow + kb * 32, 2048);
      v16bf b0 = load_frag_pe(s_nT, p0, kb * 32);
      v16bf a1 = load_frag(wrow + (kb + 1) * 32, 2048);
      v16bf b1 = load_frag_pe(s_nT, p0, (kb + 1) * 32);
      WMMA_BF16(acc0, a0, b0);
      WMMA_BF16(acc1, a1, b1);
    }
    v8f acc = acc0 + acc1;
    int cb = oct * 16 + hi * 8;
#pragma unroll
    for (int e = 0; e < 8; ++e) lds_f[px * 16 + r][cb + e] = acc[e] + pe_b[cb + e];
  }
  __syncthreads();
  if (threadIdx.x < 32) {
    float s = 0.f, s2 = 0.f;
    for (int c = 0; c < 128; ++c) { float v = lds_f[threadIdx.x][c]; s += v; s2 += v * v; }
    float m = s * (1.0f / 128.0f);
    mu[threadIdx.x] = m;
    rs[threadIdx.x] = rsqrtf(s2 * (1.0f / 128.0f) - m * m + 1e-5f);
  }
  __syncthreads();
  for (int idx = threadIdx.x; idx < 4096; idx += 256) {
    int pix = idx >> 7, c = idx & 127;
    float v = (lds_f[pix][c] - mu[pix]) * rs[pix] * ln_g[c] + ln_b[c];
    fT[(size_t)(pbase + pix) * 128 + c] = (__bf16)v;
  }
}

// ---------------------------------------------------------------------------
// K4: fe = f @ ex_w  [4096,512] + Final_PatchExpand2D + LN -> fexpT [65536,32]
// ---------------------------------------------------------------------------
__global__ __launch_bounds__(256) void k_expand(
    const __bf16* fT, const __bf16* wA_ex, const float* ln_g, const float* ln_b,
    __bf16* fexpT) {
  __shared__ float lds_fe[16][520];
  const int wave = threadIdx.x >> 5, lane = threadIdx.x & 31;
  const int r = lane & 15, hi = lane >> 4;
  const int pbase = blockIdx.x * 16;
  for (int jt = wave; jt < 32; jt += 8) {
    v8f acc0 = zero8(), acc1 = zero8();
#pragma unroll
    for (int kb = 0; kb < 4; kb += 2) {
      v16bf a0 = load_frag(wA_ex + jt * 16 * 128 + kb * 32, 128);
      v16bf b0 = load_frag(fT + (size_t)pbase * 128 + kb * 32, 128);
      v16bf a1 = load_frag(wA_ex + jt * 16 * 128 + (kb + 1) * 32, 128);
      v16bf b1 = load_frag(fT + (size_t)pbase * 128 + (kb + 1) * 32, 128);
      WMMA_BF16(acc0, a0, b0);
      WMMA_BF16(acc1, a1, b1);
    }
    v8f acc = acc0 + acc1;
    int jb = jt * 16 + hi * 8;
#pragma unroll
    for (int e = 0; e < 8; ++e) lds_fe[r][jb + e] = acc[e];
  }
  __syncthreads();
  {
    int g = threadIdx.x;
    int pix = g >> 4, d1 = (g >> 2) & 3, d2 = g & 3;
    const float* src = &lds_fe[pix][d1 * 128 + d2 * 32];
    float s = 0.f, s2 = 0.f;
    for (int c = 0; c < 32; ++c) { float v = src[c]; s += v; s2 += v * v; }
    float m = s * (1.0f / 32.0f);
    float irs = rsqrtf(s2 * (1.0f / 32.0f) - m * m + 1e-5f);
    int patch = pbase + pix;
    int b = patch >> 10, ph = (patch >> 5) & 31, pw = patch & 31;
    size_t row = (size_t)((b << 14) + ((ph * 4 + d1) << 7) + (pw * 4 + d2));
    __bf16* dst = fexpT + row * 32;
    for (int c = 0; c < 32; ++c)
      dst[c] = (__bf16)((src[c] - m) * irs * ln_g[c] + ln_b[c]);
  }
}

// ---------------------------------------------------------------------------
// K5a: xT = w0 * wf_pre(query) + w1 * (up(fexp) + up_b)   [N,128] bf16
// ---------------------------------------------------------------------------
__global__ __launch_bounds__(256) void k_wf_pre(
    const __bf16* qT, const __bf16* fexpT, const __bf16* wA_pre, const __bf16* wA_up,
    const float* up_b, const float* wf_w, __bf16* xT) {
  const int wave = threadIdx.x >> 5, lane = threadIdx.x & 31;
  const int r = lane & 15, hi = lane >> 4;
  const int pixbase = blockIdx.x * 32;
  float a0 = fmaxf(wf_w[0], 0.f), a1 = fmaxf(wf_w[1], 0.f);
  float inv = fast_rcp(a0 + a1 + 1e-8f);
  float w0 = a0 * inv, w1 = a1 * inv;
  for (int t = wave; t < 16; t += 8) {
    int px = t & 1, oct = t >> 1;
    int n0 = pixbase + px * 16;
    v8f accP0 = zero8(), accP1 = zero8(), accU = zero8();
#pragma unroll
    for (int kb = 0; kb < 4; kb += 2) {
      v16bf a0f = load_frag(wA_pre + oct * 16 * 128 + kb * 32, 128);
      v16bf b0f = load_frag(qT + (size_t)n0 * 128 + kb * 32, 128);
      v16bf a1f = load_frag(wA_pre + oct * 16 * 128 + (kb + 1) * 32, 128);
      v16bf b1f = load_frag(qT + (size_t)n0 * 128 + (kb + 1) * 32, 128);
      WMMA_BF16(accP0, a0f, b0f);
      WMMA_BF16(accP1, a1f, b1f);
    }
    {
      v16bf a  = load_frag(wA_up + oct * 16 * 32, 32);
      v16bf bf = load_frag(fexpT + (size_t)n0 * 32, 32);
      WMMA_BF16(accU, a, bf);
    }
    v8f accP = accP0 + accP1;
    int cb = oct * 16 + hi * 8;
    v8bf pk;
#pragma unroll
    for (int e = 0; e < 8; ++e)
      pk[e] = (__bf16)(w0 * accP[e] + w1 * (accU[e] + up_b[cb + e]));
    *(v8bf*)(xT + (size_t)(n0 + r) * 128 + cb) = pk;
  }
}

// ---------------------------------------------------------------------------
// K5b: 3x3 SAME conv (K=1152) + BN + ReLU6 -> updated_query (NCHW f32 d_out);
// fused write-phase: per-pixel l2norm + class-wise scatter sums (LDS->global).
// ---------------------------------------------------------------------------
__global__ __launch_bounds__(256) void k_wf_post(
    const __bf16* xT, const __bf16* wA_post, const float* bn_g, const float* bn_b,
    const int* mask, float* out_q, float* gfs, float* gcnt) {
  __shared__ float lds_uq[32][132];
  __shared__ float rstd[32];
  __shared__ int   kcls[32];
  __shared__ float lfs[768];
  __shared__ float lcnt[6];
  const int wave = threadIdx.x >> 5, lane = threadIdx.x & 31;
  const int r = lane & 15, hi = lane >> 4;
  const int pixbase = blockIdx.x * 32;
  for (int i = threadIdx.x; i < 774; i += 256) {
    if (i < 768) lfs[i] = 0.f; else lcnt[i - 768] = 0.f;
  }
  const float bnscale = rsqrtf(1.0f + 1e-5f);   // running var 1.0
  for (int t = wave; t < 16; t += 8) {
    int px = t & 1, oct = t >> 1;
    int n0 = pixbase + px * 16;
    const __bf16* wrow = wA_post + oct * 16 * 1152;
    v8f acc0 = zero8(), acc1 = zero8();
    for (int kb = 0; kb < 36; kb += 2) {
      if ((kb & 7) == 0)                       // gfx1250 prefetch of A stream
        __builtin_prefetch(wrow + (kb + 8) * 32, 0, 0);
      v16bf a0 = load_frag(wrow + kb * 32, 1152);
      v16bf b0 = load_frag_3x3(xT, n0, kb * 32);
      v16bf a1 = load_frag(wrow + (kb + 1) * 32, 1152);
      v16bf b1 = load_frag_3x3(xT, n0, (kb + 1) * 32);
      WMMA_BF16(acc0, a0, b0);
      WMMA_BF16(acc1, a1, b1);
    }
    v8f acc = acc0 + acc1;
    int cb = oct * 16 + hi * 8;
    int n = n0 + r;
    int b = n >> 14, pix = n & 16383;
#pragma unroll
    for (int e = 0; e < 8; ++e) {
      int c = cb + e;
      float v = acc[e] * (bn_g[c] * bnscale) + bn_b[c];
      v = fminf(fmaxf(v, 0.0f), 6.0f);          // ReLU6
      out_q[(size_t)b * 2097152 + (size_t)c * 16384 + pix] = v;
      lds_uq[px * 16 + r][c] = v;
    }
  }
  __syncthreads();
  if (threadIdx.x < 32) {
    float ss = 0.f;
    for (int c = 0; c < 128; ++c) { float v = lds_uq[threadIdx.x][c]; ss += v * v; }
    rstd[threadIdx.x] = fast_rcp(fmaxf(sqrtf(ss), 1e-12f));
    int k = mask[pixbase + threadIdx.x];
    kcls[threadIdx.x] = k;
    atomicAdd(&lcnt[k], 1.0f);
  }
  __syncthreads();
  for (int idx = threadIdx.x; idx < 4096; idx += 256) {
    int pix = idx >> 7, c = idx & 127;
    atomicAdd(&lfs[c * 6 + kcls[pix]], lds_uq[pix][c] * rstd[pix]);
  }
  __syncthreads();
  for (int i = threadIdx.x; i < 768; i += 256) atomicAdd(&gfs[i], lfs[i]);
  if (threadIdx.x < 6) atomicAdd(&gcnt[threadIdx.x], lcnt[threadIdx.x]);
}

// ---------------------------------------------------------------------------
// K6: updated_memory = 0.7 * l2norm(m_items) + 0.3 * (feat_sum/cnt)[:M]
// ---------------------------------------------------------------------------
__global__ void k_memory(const float* m_items, const float* gfs, const float* gcnt,
                         float* out_mem) {
  __shared__ float rn[5];
  if (threadIdx.x < 5) {
    float ss = 0.f;
    for (int c = 0; c < 128; ++c) { float v = m_items[threadIdx.x * 128 + c]; ss += v * v; }
    rn[threadIdx.x] = fast_rcp(fmaxf(sqrtf(ss), 1e-12f));
  }
  __syncthreads();
  for (int i = threadIdx.x; i < 640; i += 256) {
    int m = i >> 7, c = i & 127;
    float nm = gfs[c * 6 + m] / (gcnt[m] + 1e-8f);
    out_mem[i] = 0.7f * m_items[i] * rn[m] + 0.3f * nm;
  }
}

// ---------------------------------------------------------------------------
extern "C" void kernel_launch(void* const* d_in, const int* in_sizes, int n_in,
                              void* d_out, int out_size, void* d_ws, size_t ws_size,
                              hipStream_t stream) {
  (void)in_sizes; (void)n_in; (void)out_size; (void)ws_size;
  const float* S       = (const float*)d_in[0];
  const float* Q       = (const float*)d_in[1];
  const int*   mask    = (const int*)  d_in[2];
  const float* sm_w    = (const float*)d_in[3];
  const float* sm_b    = (const float*)d_in[4];
  const float* c1_w    = (const float*)d_in[5];
  const float* c1_b    = (const float*)d_in[6];
  const float* c2_w    = (const float*)d_in[7];
  const float* c2_b    = (const float*)d_in[8];
  const float* pe_w    = (const float*)d_in[9];
  const float* pe_b    = (const float*)d_in[10];
  const float* pe_ln_g = (const float*)d_in[11];
  const float* pe_ln_b = (const float*)d_in[12];
  const float* ex_w    = (const float*)d_in[13];
  const float* ex_ln_g = (const float*)d_in[14];
  const float* ex_ln_b = (const float*)d_in[15];
  const float* up_w    = (const float*)d_in[16];
  const float* up_b    = (const float*)d_in[17];
  const float* pre_w   = (const float*)d_in[18];
  const float* wf_w    = (const float*)d_in[19];
  const float* post_w  = (const float*)d_in[20];
  const float* bn_g    = (const float*)d_in[21];
  const float* bn_b    = (const float*)d_in[22];
  const float* m_items = (const float*)d_in[23];

  char* ws = (char*)d_ws;                       // needs ~73.7 MB
  __bf16* snT    = (__bf16*)(ws + 0);           // [65536,128] bf16
  __bf16* qT     = (__bf16*)(ws + 16777216);    // [65536,128] bf16
  __bf16* s_nT   = (__bf16*)(ws + 33554432);    // [65536,128] bf16
  __bf16* fexpT  = (__bf16*)(ws + 50331648);    // [65536, 32] bf16
  __bf16* fT     = (__bf16*)(ws + 54525952);    // [4096,128]  bf16
  __bf16* xT     = (__bf16*)(ws + 55574528);    // [65536,128] bf16
  float*  accb   = (float*) (ws + 72351744);    // feat_sum[768] + cnt[6]
  __bf16* wb     = (__bf16*)(ws + 72355840);    // bf16 weights
  __bf16* wA_sm  = wb;
  __bf16* wA_c1  = wb + 81920;
  __bf16* wA_c2  = wb + 122880;
  __bf16* wA_pe  = wb + 131072;
  __bf16* wA_ex  = wb + 393216;
  __bf16* wA_up  = wb + 458752;
  __bf16* wA_pre = wb + 462848;
  __bf16* wA_post= wb + 479232;
  float* gfs  = accb;
  float* gcnt = accb + 768;
  float* out_q   = (float*)d_out;
  float* out_mem = out_q + 8388608;

  k_prep_weights <<<2448, 256, 0, stream>>>(sm_w, c1_w, c2_w, pe_w, ex_w, up_w, pre_w, post_w, wb);
  k_zero_acc     <<<1,    256, 0, stream>>>(accb);
  k_norm_transpose<<<256, 256, 0, stream>>>(S, Q, snT, qT);
  k_read_fused   <<<2048, 256, 0, stream>>>(snT, wA_sm, wA_c1, wA_c2, sm_b, c1_b, c2_b, m_items, s_nT);
  k_patch_embed  <<<128,  256, 0, stream>>>(s_nT, wA_pe, pe_b, pe_ln_g, pe_ln_b, fT);
  k_expand       <<<256,  256, 0, stream>>>(fT, wA_ex, ex_ln_g, ex_ln_b, fexpT);
  k_wf_pre       <<<2048, 256, 0, stream>>>(qT, fexpT, wA_pre, wA_up, up_b, wf_w, xT);
  k_wf_post      <<<2048, 256, 0, stream>>>(xT, wA_post, bn_g, bn_b, mask, out_q, gfs, gcnt);
  k_memory       <<<1,    256, 0, stream>>>(m_items, gfs, gcnt, out_mem);
}
